// DecisionMaking_35562329211645
// MI455X (gfx1250) — matrix operand order repaired and checked by reading, not compiled, not run on hardware
//
#include <hip/hip_runtime.h>
#include <hip/hip_bf16.h>
#include <math.h>

// Problem constants (match reference)
#define Bv 8
#define Nv 512
#define Mv 64
#define Dv 128
#define Hv 64
#define NMv (Nv * Mv) // 32768

typedef __attribute__((ext_vector_type(16))) __bf16 v16bf;
typedef __attribute__((ext_vector_type(8)))  float  v8f;

// float -> bf16 fptrunc (RNE) -> native v_cvt_pk_bf16_f32 on gfx1250
static __device__ __forceinline__ __bf16 f2bf(float f) {
    return static_cast<__bf16>(f);
}

static __device__ __forceinline__ v16bf pack16(float4 x0, float4 x1, float4 x2, float4 x3) {
    v16bf a;
    a[0]  = f2bf(x0.x); a[1]  = f2bf(x0.y); a[2]  = f2bf(x0.z); a[3]  = f2bf(x0.w);
    a[4]  = f2bf(x1.x); a[5]  = f2bf(x1.y); a[6]  = f2bf(x1.z); a[7]  = f2bf(x1.w);
    a[8]  = f2bf(x2.x); a[9]  = f2bf(x2.y); a[10] = f2bf(x2.z); a[11] = f2bf(x2.w);
    a[12] = f2bf(x3.x); a[13] = f2bf(x3.y); a[14] = f2bf(x3.z); a[15] = f2bf(x3.w);
    return a;
}

// ---------------------------------------------------------------------------
// Generic first-layer WMMA GEMM: Y[r, h] = X[r, 0:128] @ W1[wrow0:wrow0+128, h]
//                                          (+ bias[h] if bias != nullptr)
// block = 256 threads = 8 waves, 16 rows per wave; rows must be mult. of 128.
// ---------------------------------------------------------------------------
__global__ __launch_bounds__(256) void emb_gemm_kernel(
    const float* __restrict__ X, const float* __restrict__ W1, int wrow0,
    const float* __restrict__ bias, float* __restrict__ Y)
{
    // WMMA B-layout (16-bit, 32x16): lane l holds col = l&15, K = (l>>4)*8 + e + (e&8)
    __shared__ __align__(32) __bf16 WB[4][4][32][16];
    __shared__ float sbias[64];

    const int tid = threadIdx.x;
    #pragma unroll
    for (int i = 0; i < 2; ++i) {
        int c    = tid + i * 256;
        int kk   = c >> 7;
        int rest = c & 127;
        int nt   = rest >> 5;
        int l    = rest & 31;
        int col  = nt * 16 + (l & 15);
        int Kb   = kk * 32 + (l >> 4) * 8;
        #pragma unroll
        for (int e = 0; e < 16; ++e)
            WB[kk][nt][l][e] = f2bf(W1[(wrow0 + Kb + e + (e & 8)) * Hv + col]);
    }
    if (tid < 64) sbias[tid] = bias ? bias[tid] : 0.0f;
    __syncthreads();

    const int w     = tid >> 5;
    const int lane  = tid & 31;
    const int rowa  = lane & 15;
    const int kgrp  = lane >> 4;
    const int kbase = kgrp * 8;

    const int r0 = (blockIdx.x * 8 + w) * 16;

    v8f acc[4];
    #pragma unroll
    for (int nt = 0; nt < 4; ++nt) {
        float bb = sbias[nt * 16 + rowa];
        #pragma unroll
        for (int v = 0; v < 8; ++v) acc[nt][v] = bb;
    }

    const float* xr = X + (size_t)(r0 + rowa) * Dv;
    #pragma unroll
    for (int kk = 0; kk < 4; ++kk) {
        int K0 = kk * 32 + kbase;
        float4 x0 = *(const float4*)(xr + K0);
        float4 x1 = *(const float4*)(xr + K0 + 4);
        float4 x2 = *(const float4*)(xr + K0 + 16);
        float4 x3 = *(const float4*)(xr + K0 + 20);
        v16bf a = pack16(x0, x1, x2, x3);
        v16bf bm[4];
        #pragma unroll
        for (int nt = 0; nt < 4; ++nt)
            bm[nt] = *(const v16bf*)(&WB[kk][nt][lane][0]);
        #pragma unroll
        for (int nt = 0; nt < 4; ++nt)
            acc[nt] = __builtin_amdgcn_wmma_f32_16x16x32_bf16(
                false, a, false, bm[nt], (short)0, acc[nt], false, false);
    }

    // C-layout writeback: VGPR v holds row v + 8*(lane>=16), col = nt*16+(lane&15)
    #pragma unroll
    for (int nt = 0; nt < 4; ++nt) {
        int col = nt * 16 + rowa;
        #pragma unroll
        for (int v = 0; v < 8; ++v) {
            int row = v + 8 * kgrp;
            Y[(size_t)(r0 + row) * Hv + col] = acc[nt][v];
        }
    }
}

// ---------------------------------------------------------------------------
// Main kernel: per wave, FOUR consecutive 16-action tiles (one full (b,n)
// over all m) through the 3-layer MLP via WMMA.
// block = 256 threads = 8 waves; grid = (B*N*M/16)/(8*4) = 512 blocks.
// ---------------------------------------------------------------------------
__global__ __launch_bounds__(256) void score_kernel(
    const float* __restrict__ edge,    // [B*N*M, 128]
    const float* __restrict__ Aop,     // [B*N, 64] (b1 folded in)
    const float* __restrict__ Amach,   // [B*M, 64]
    const float* __restrict__ W1,      // [384, 64]; We = rows 256..383
    const float* __restrict__ W2,      // [64, 64]
    const float* __restrict__ W3,      // [64]
    const float* __restrict__ b2,      // [64]
    const float* __restrict__ b3,      // [1]
    const int*   __restrict__ mask,    // [B*N*M]
    float* __restrict__ scores)        // [B*N*M]
{
    __shared__ __align__(32) __bf16 WeB[4][4][32][16];  // [kslab][ntile][lane][e] 16KB
    __shared__ __align__(32) __bf16 W2B[2][4][32][16];  // 8KB
    __shared__ float sW3[64];
    __shared__ float sb2[64];
    __shared__ float h1s[8][16 * 68];                   // per-wave scratch, padded stride 68

    const int tid = threadIdx.x;

    // Fill WeB: 512 lane-chunks, 2 per thread
    #pragma unroll
    for (int i = 0; i < 2; ++i) {
        int c    = tid + i * 256;
        int kk   = c >> 7;
        int rest = c & 127;
        int nt   = rest >> 5;
        int l    = rest & 31;
        int col  = nt * 16 + (l & 15);
        int Kb   = kk * 32 + (l >> 4) * 8;
        #pragma unroll
        for (int e = 0; e < 16; ++e)
            WeB[kk][nt][l][e] = f2bf(W1[(2 * Dv + Kb + e + (e & 8)) * Hv + col]);
    }
    // Fill W2B: 256 lane-chunks, 1 per thread
    {
        int c   = tid;
        int kk  = c >> 7;
        int nt  = (c >> 5) & 3;
        int l   = c & 31;
        int col = nt * 16 + (l & 15);
        int Kb  = kk * 32 + (l >> 4) * 8;
        #pragma unroll
        for (int e = 0; e < 16; ++e)
            W2B[kk][nt][l][e] = f2bf(W2[(Kb + e + (e & 8)) * Hv + col]);
    }
    if (tid < 64) { sW3[tid] = W3[tid]; sb2[tid] = b2[tid]; }
    __syncthreads();

    const int w     = tid >> 5;
    const int lane  = tid & 31;
    const int rowa  = lane & 15;   // A-matrix row / C-matrix column-in-tile
    const int kgrp  = lane >> 4;
    const int kbase = kgrp * 8;    // A-layout K split for 16-bit data

    const int t0  = (blockIdx.x * 8 + w) * 4;  // 4 consecutive tiles per wave
    const int r00 = t0 * 16;                   // 64 rows = all m of one (b,n)
    const int b   = r00 >> 15;
    const int bn  = r00 >> 6;                  // b*N + n

    // Aop is identical for all 4 tiles of this wave: hoist.
    float aopv[4];
    #pragma unroll
    for (int nt = 0; nt < 4; ++nt) aopv[nt] = Aop[bn * Hv + nt * 16 + rowa];

    float* hw = &h1s[w][0];

    for (int it = 0; it < 4; ++it) {
        const int r0 = r00 + it * 16;
        const int m0 = it * 16;

        // ---- Layer 1: C init = Aop (bcast over rows) + Amach (per m-row) ----
        v8f acc[4];
        #pragma unroll
        for (int nt = 0; nt < 4; ++nt) {
            int col = nt * 16 + rowa;
            #pragma unroll
            for (int v = 0; v < 8; ++v) {
                int mrow = m0 + v + 8 * kgrp;       // C-layout row
                acc[nt][v] = aopv[nt] + Amach[(b * Mv + mrow) * Hv + col];
            }
        }

        // edge A-tile: 16 rows x 128 K, bf16 A-layout
        const float* erow = edge + (size_t)(r0 + rowa) * Dv;
        #pragma unroll
        for (int kk = 0; kk < 4; ++kk) {
            int K0 = kk * 32 + kbase;
            float4 x0 = *(const float4*)(erow + K0);
            float4 x1 = *(const float4*)(erow + K0 + 4);
            float4 x2 = *(const float4*)(erow + K0 + 16);
            float4 x3 = *(const float4*)(erow + K0 + 20);
            v16bf a = pack16(x0, x1, x2, x3);
            v16bf bm[4];
            #pragma unroll
            for (int nt = 0; nt < 4; ++nt)
                bm[nt] = *(const v16bf*)(&WeB[kk][nt][lane][0]);
            #pragma unroll
            for (int nt = 0; nt < 4; ++nt)
                acc[nt] = __builtin_amdgcn_wmma_f32_16x16x32_bf16(
                    false, a, false, bm[nt], (short)0, acc[nt], false, false);
        }

        // ReLU + D->A relayout bounce through LDS (stride 68 => conflict-free)
        #pragma unroll
        for (int nt = 0; nt < 4; ++nt) {
            int col = nt * 16 + rowa;
            #pragma unroll
            for (int v = 0; v < 8; ++v) {
                int row = v + 8 * kgrp;
                hw[row * 68 + col] = fmaxf(acc[nt][v], 0.0f);
            }
        }
        asm volatile("s_wait_dscnt 0" ::: "memory"); // wave-internal RAW via LDS

        // ---- Layer 2: h1(16x64) @ W2(64x64) + b2 ----
        v8f acc2[4];
        #pragma unroll
        for (int nt = 0; nt < 4; ++nt) {
            float bb = sb2[nt * 16 + rowa];
            #pragma unroll
            for (int v = 0; v < 8; ++v) acc2[nt][v] = bb;
        }
        const float* hr = hw + rowa * 68;
        #pragma unroll
        for (int kk = 0; kk < 2; ++kk) {
            int K0 = kk * 32 + kbase;
            float4 y0 = *(const float4*)(hr + K0);
            float4 y1 = *(const float4*)(hr + K0 + 4);
            float4 y2 = *(const float4*)(hr + K0 + 16);
            float4 y3 = *(const float4*)(hr + K0 + 20);
            v16bf a2 = pack16(y0, y1, y2, y3);
            v16bf bm[4];
            #pragma unroll
            for (int nt = 0; nt < 4; ++nt)
                bm[nt] = *(const v16bf*)(&W2B[kk][nt][lane][0]);
            #pragma unroll
            for (int nt = 0; nt < 4; ++nt)
                acc2[nt] = __builtin_amdgcn_wmma_f32_16x16x32_bf16(
                    false, a2, false, bm[nt], (short)0, acc2[nt], false, false);
        }

        // ReLU h2 -> same LDS scratch
        #pragma unroll
        for (int nt = 0; nt < 4; ++nt) {
            int col = nt * 16 + rowa;
            #pragma unroll
            for (int v = 0; v < 8; ++v) {
                int row = v + 8 * kgrp;
                hw[row * 68 + col] = fmaxf(acc2[nt][v], 0.0f);
            }
        }
        asm volatile("s_wait_dscnt 0" ::: "memory");

        // ---- Layer 3: score = h2 @ W3 + b3, masked ----
        int cb = kgrp * 32;
        float s = 0.f;
        #pragma unroll
        for (int c = 0; c < 32; ++c) s += hw[rowa * 68 + cb + c] * sW3[cb + c];
        s += __shfl_xor(s, 16, 32);
        if (lane < 16) {
            int r = r0 + rowa;
            float sc = s + b3[0];
            scores[r] = mask[r] ? sc : -INFINITY;
        }
    }
}

// ---------------------------------------------------------------------------
// Per-batch masked log-softmax over 32768 scores. 1 block/batch, 1024 threads,
// 32 elements per thread held in registers across both passes.
// ---------------------------------------------------------------------------
__global__ __launch_bounds__(1024) void logsoftmax_kernel(
    const float* __restrict__ scores, float* __restrict__ out)
{
    __shared__ float red[32];
    const int b   = blockIdx.x;
    const int tid = threadIdx.x;
    const float* s = scores + (size_t)b * NMv;
    float v[32];
    #pragma unroll
    for (int j = 0; j < 32; ++j) v[j] = s[tid + j * 1024];

    // max reduction
    float mx = -INFINITY;
    #pragma unroll
    for (int j = 0; j < 32; ++j) mx = fmaxf(mx, v[j]);
    #pragma unroll
    for (int off = 16; off > 0; off >>= 1) mx = fmaxf(mx, __shfl_xor(mx, off, 32));
    if ((tid & 31) == 0) red[tid >> 5] = mx;
    __syncthreads();
    if (tid < 32) {
        float m = red[tid];
        #pragma unroll
        for (int off = 16; off > 0; off >>= 1) m = fmaxf(m, __shfl_xor(m, off, 32));
        red[tid] = m;
    }
    __syncthreads();
    mx = red[0];
    __syncthreads();

    // sum(exp(x - max)) reduction
    float se = 0.f;
    #pragma unroll
    for (int j = 0; j < 32; ++j) se += __expf(v[j] - mx);
    #pragma unroll
    for (int off = 16; off > 0; off >>= 1) se += __shfl_xor(se, off, 32);
    if ((tid & 31) == 0) red[tid >> 5] = se;
    __syncthreads();
    if (tid < 32) {
        float m = red[tid];
        #pragma unroll
        for (int off = 16; off > 0; off >>= 1) m += __shfl_xor(m, off, 32);
        red[tid] = m;
    }
    __syncthreads();
    float lse = mx + logf(red[0]);

    float* o = out + (size_t)b * NMv;
    #pragma unroll
    for (int j = 0; j < 32; ++j) o[tid + j * 1024] = v[j] - lse;
}

// ---------------------------------------------------------------------------
extern "C" void kernel_launch(void* const* d_in, const int* in_sizes, int n_in,
                              void* d_out, int out_size, void* d_ws, size_t ws_size,
                              hipStream_t stream)
{
    const float* op   = (const float*)d_in[0];
    const float* mach = (const float*)d_in[1];
    const float* edge = (const float*)d_in[2];
    const int*   mask = (const int*)d_in[3];
    const float* W1   = (const float*)d_in[4];
    const float* b1   = (const float*)d_in[5];
    const float* W2   = (const float*)d_in[6];
    const float* b2   = (const float*)d_in[7];
    const float* W3   = (const float*)d_in[8];
    const float* b3   = (const float*)d_in[9];
    float* out = (float*)d_out;

    char* ws = (char*)d_ws;
    float* scores = (float*)ws;                                              // B*N*M = 262144 f
    float* Aop    = (float*)(ws + (size_t)262144 * 4);                       // B*N*H = 262144 f (2097152 f reserved)
    float* Amach  = (float*)(ws + (size_t)262144 * 4 + (size_t)2097152 * 4); // B*M*H = 32768 f

    // First-layer partial GEMMs via WMMA (rows multiple of 128 in both cases)
    emb_gemm_kernel<<<(Bv * Nv) / 128, 256, 0, stream>>>(op,   W1, 0,  b1,      Aop);
    emb_gemm_kernel<<<(Bv * Mv) / 128, 256, 0, stream>>>(mach, W1, Dv, nullptr, Amach);
    score_kernel   <<<(Bv * Nv * Mv) / (16 * 8 * 4), 256, 0, stream>>>(
        edge, Aop, Amach, W1, W2, W3, b2, b3, mask, scores);
    logsoftmax_kernel<<<Bv, 1024, 0, stream>>>(scores, out);
}
